// WaveNet_15745350107430
// MI455X (gfx1250) — matrix operand order repaired and checked by reading, not compiled, not run on hardware
//
#include <hip/hip_runtime.h>
#include <hip/hip_bf16.h>

#define TDIM   16384
#define BATCH  4
#define ROWS   (BATCH*TDIM)        // 65536 tokens
#define CCH    128
#define NCONV  256
#define KPAD   136                  // padded K stride (halves) -> conflict-free LDS
#define LAYERS 30

typedef __attribute__((ext_vector_type(16))) _Float16 v16h;
typedef __attribute__((ext_vector_type(8)))  float    v8f;

#if defined(__has_builtin)
#if __has_builtin(__builtin_amdgcn_sched_group_barrier)
#define SGB(mask, size) __builtin_amdgcn_sched_group_barrier((mask), (size), 0)
#endif
#endif
#ifndef SGB
#define SGB(mask, size)
#endif
#define SGB_WMMA 0x008   // MFMA/WMMA sched class
#define SGB_DSRD 0x100   // DS read sched class

// ---- fragment loaders -------------------------------------------------------
// A-operand layout (16-bit A 16x32): lane holds K in {b..b+7} U {b+16..b+23}, b=8*(lane/16)
__device__ __forceinline__ v16h load_a_lds(const _Float16* p) {
  v16h a;
  *((uint4*)&a)       = *((const uint4*)p);
  *(((uint4*)&a) + 1) = *((const uint4*)(p + 16));
  return a;
}
// B-operand layout (16-bit B 32x16): lane holds K in {b..b+7} U {b+16..b+23}, b=8*(lane/16)
__device__ __forceinline__ v16h load_b_lds(const _Float16* p) {
  v16h b;
  *((uint4*)&b)       = *((const uint4*)p);
  *(((uint4*)&b) + 1) = *((const uint4*)(p + 16));
  return b;
}

__device__ __forceinline__ v8f wmma_f16(v16h a, v16h b, v8f c) {
  return __builtin_amdgcn_wmma_f32_16x16x32_f16(false, a, false, b, (short)0, c,
                                                false, false);
}

// ---- weight prep: transpose + f16 + pad -------------------------------------
// blk_conv_W [L][3][128][256] -> convT [L][3][256][KPAD] f16 ([co][ci])
__global__ void wn_prep_conv(const float* __restrict__ W, _Float16* __restrict__ out) {
  size_t idx = (size_t)blockIdx.x * 256 + threadIdx.x;
  const size_t total = (size_t)LAYERS * 3 * NCONV * KPAD;
  if (idx >= total) return;
  int ci = (int)(idx % KPAD);
  size_t t = idx / KPAD;
  int co = (int)(t % NCONV); t /= NCONV;
  int tap = (int)(t % 3);
  int l   = (int)(t / 3);
  float v = (ci < CCH) ? W[(((size_t)(l*3 + tap)*CCH + ci))*NCONV + co] : 0.f;
  out[idx] = (_Float16)v;
}

// [L][128][128] ([ci][co]) -> [L][128][KPAD] ([co][ci]) f16
__global__ void wn_prep_cc(const float* __restrict__ W, _Float16* __restrict__ out) {
  size_t idx = (size_t)blockIdx.x * 256 + threadIdx.x;
  const size_t total = (size_t)LAYERS * CCH * KPAD;
  if (idx >= total) return;
  int ci = (int)(idx % KPAD);
  size_t t = idx / KPAD;
  int co = (int)(t % CCH);
  int l  = (int)(t / CCH);
  float v = (ci < CCH) ? W[((size_t)l*CCH + ci)*CCH + co] : 0.f;
  out[idx] = (_Float16)v;
}

// ---- timestep embedding + swish MLP (tiny; one block) -----------------------
__global__ void wn_embed(const int* __restrict__ timestep,
                         const float* __restrict__ W0, const float* __restrict__ b0,
                         const float* __restrict__ W1, const float* __restrict__ b1,
                         float* __restrict__ embedOut) {
  __shared__ float e0[BATCH * CCH];   // sin/cos embedding
  __shared__ float h1[BATCH * 512];
  const int tid = threadIdx.x;        // 512 threads
  if (tid < BATCH * CCH) {
    int b = tid >> 7, c = tid & 127;
    float tval = (float)timestep[b];
    int j = c & 63;
    float expo = powf(10.f, ((float)j / 63.f) * 4.f);
    float comp = expo * tval;
    e0[tid] = (c < 64) ? sinf(comp) : cosf(comp);
  }
  __syncthreads();
  for (int idx = tid; idx < BATCH * 512; idx += 512) {
    int b = idx >> 9, p = idx & 511;
    float acc = b0[p];
    for (int q = 0; q < CCH; ++q) acc += e0[b*CCH + q] * W0[q*512 + p];
    h1[idx] = acc / (1.f + __expf(-acc));
  }
  __syncthreads();
  for (int idx = tid; idx < BATCH * 512; idx += 512) {
    int b = idx >> 9, p = idx & 511;
    float acc = b1[p];
    for (int q = 0; q < 512; ++q) acc += h1[b*512 + q] * W1[q*512 + p];
    embedOut[idx] = acc / (1.f + __expf(-acc));
  }
}

// e_all[l][b][c] = embed[b] . blk_embed_W[l][:,c] + blk_embed_b[l][c]
__global__ void wn_eall(const float* __restrict__ embed, const float* __restrict__ eW,
                        const float* __restrict__ eb, float* __restrict__ e_all) {
  int idx = blockIdx.x * 256 + threadIdx.x;
  if (idx >= LAYERS * BATCH * CCH) return;
  int l = idx / (BATCH * CCH);
  int rem = idx & (BATCH * CCH - 1);
  int b = rem >> 7, c = rem & 127;
  float acc = eb[l*CCH + c];
  const float* W  = eW + (size_t)l * 512 * CCH;
  const float* em = embed + b * 512;
  for (int q = 0; q < 512; ++q) acc += em[q] * W[q*CCH + c];
  e_all[(size_t)idx] = acc;
}

// input 1x1 conv + relu, zero skip
__global__ void wn_init(const float* __restrict__ signal, const float* __restrict__ inW,
                        const float* __restrict__ inb, float* __restrict__ x,
                        float* __restrict__ skip) {
  size_t idx = (size_t)blockIdx.x * 256 + threadIdx.x;   // ROWS*CCH
  size_t r = idx >> 7; int c = (int)(idx & 127);
  float v = signal[r] * inW[c] + inb[c];
  x[idx]    = v > 0.f ? v : 0.f;
  skip[idx] = 0.f;
}

// ---- fused WMMA residual block ---------------------------------------------
// LDS: lds_w (68KB, conv tap weights -> later res+skip weights) + lds_g (34KB)
// = 102KB total => 3 workgroups per WGP.
__global__ __launch_bounds__(256) void wn_layer(
    const float* __restrict__ xin, float* __restrict__ xout, float* __restrict__ skip,
    const _Float16* __restrict__ convT,  // [3][256][KPAD] this layer
    const _Float16* __restrict__ resT,   // [128][KPAD]
    const _Float16* __restrict__ skipT,  // [128][KPAD]
    const float* __restrict__ conv_b, const float* __restrict__ res_b,
    const float* __restrict__ skip_b, const float* __restrict__ e, int dil) {
  __shared__ _Float16 lds_w[NCONV * KPAD];       // 69,632 B
  __shared__ _Float16 lds_g[8 * 16 * KPAD];      // 34,816 B

  const int tid   = threadIdx.x;
  const int lane  = tid & 31;
  const int wave  = tid >> 5;                    // 0..7
  const int mrow  = lane & 15;
  const int khalf = lane >> 4;                   // 0 or 1
  const int r0    = blockIdx.x * 128 + wave * 16;
  const int rA    = r0 + mrow;
  const int batch = rA >> 14;
  const int tloc  = rA & (TDIM - 1);

  const v8f vzero = {0.f,0.f,0.f,0.f,0.f,0.f,0.f,0.f};
  v8f acc[16];
  #pragma unroll
  for (int j = 0; j < 16; ++j) acc[j] = vzero;

  const float* erow = e + batch * CCH;

  // ---- dilated conv: 3 taps staged through lds_w -------------------------
  for (int tap = 0; tap < 3; ++tap) {
    __syncthreads();
    {
      const uint4* s = (const uint4*)(convT + tap * NCONV * KPAD);
      uint4* d = (uint4*)lds_w;
      for (int i2 = tid; i2 < NCONV * KPAD / 8; i2 += 256) d[i2] = s[i2];
    }
    __syncthreads();
    const int toff = tloc + (tap - 1) * dil;
    const bool valid = ((unsigned)toff < (unsigned)TDIM);
    const float* arow = xin + ((size_t)batch * TDIM + (valid ? toff : 0)) * CCH;
    #pragma unroll
    for (int kk = 0; kk < 4; ++kk) {
      const int abase = kk * 32 + khalf * 8;
      v16h a;
      #pragma unroll
      for (int t2 = 0; t2 < 8; ++t2) {
        float lo = valid ? (arow[abase + t2]      + erow[abase + t2])      : 0.f;
        float hi = valid ? (arow[abase + 16 + t2] + erow[abase + 16 + t2]) : 0.f;
        a[t2]     = (_Float16)lo;
        a[t2 + 8] = (_Float16)hi;
      }
      const _Float16* brow = &lds_w[mrow * KPAD + kk * 32 + khalf * 8];
      #pragma unroll
      for (int nt = 0; nt < 16; ++nt) {
        v16h b = load_b_lds(brow + nt * 16 * KPAD);
        acc[nt] = wmma_f16(a, b, acc[nt]);
      }
      // forced software pipeline: keep 2 B fragments in flight
      SGB(SGB_DSRD, 4);                      // prefetch fragments 0,1
      #pragma unroll
      for (int it = 0; it < 14; ++it) {
        SGB(SGB_WMMA, 1);                    // wmma(it)
        SGB(SGB_DSRD, 2);                    // prefetch fragment it+2
      }
      SGB(SGB_WMMA, 2);                      // drain
    }
  }

  // ---- bias + gated activation: g = tanh(h[:,:128]) * sigmoid(h[:,128:]) --
  v8f g[8];
  #pragma unroll
  for (int j = 0; j < 8; ++j) {
    const float bt = conv_b[j * 16 + mrow];
    const float bs = conv_b[CCH + j * 16 + mrow];
    #pragma unroll
    for (int r = 0; r < 8; ++r) {
      float th = tanhf(acc[j][r] + bt);
      float sg = 1.f / (1.f + __expf(-(acc[j + 8][r] + bs)));
      g[j][r] = th * sg;
    }
  }

  __syncthreads();   // all waves done reading conv weights from lds_w
  { // restage res (first half) + skip (second half) into lds_w
    const uint4* s0 = (const uint4*)resT;
    const uint4* s1 = (const uint4*)skipT;
    uint4* d = (uint4*)lds_w;
    const int n4 = CCH * KPAD / 8;     // 2176 uint4 per matrix
    for (int i2 = tid; i2 < n4; i2 += 256) { d[i2] = s0[i2]; d[n4 + i2] = s1[i2]; }
  }
  // write g (D-layout -> row-major f16) into private lds_g region
  _Float16* gbase = lds_g + wave * (16 * KPAD);
  const int mw = khalf * 8;
  #pragma unroll
  for (int j = 0; j < 8; ++j)
    #pragma unroll
    for (int r = 0; r < 8; ++r)
      gbase[(r + mw) * KPAD + j * 16 + mrow] = (_Float16)g[j][r];
  __syncthreads();

  // ---- res / skip GEMMs ---------------------------------------------------
  v8f racc[8], sacc[8];
  #pragma unroll
  for (int j = 0; j < 8; ++j) { racc[j] = vzero; sacc[j] = vzero; }

  const _Float16* resL  = lds_w;
  const _Float16* skipL = lds_w + CCH * KPAD;
  #pragma unroll
  for (int kk = 0; kk < 4; ++kk) {
    v16h a = load_a_lds(gbase + mrow * KPAD + kk * 32 + khalf * 8);
    const int boff = mrow * KPAD + kk * 32 + khalf * 8;
    #pragma unroll
    for (int nt = 0; nt < 8; ++nt) {
      v16h br = load_b_lds(resL  + boff + nt * 16 * KPAD);
      v16h bs = load_b_lds(skipL + boff + nt * 16 * KPAD);
      racc[nt] = wmma_f16(a, br, racc[nt]);
      sacc[nt] = wmma_f16(a, bs, sacc[nt]);
    }
    // pipeline: A frag + first (res,skip) pair ahead, then 2 wmma : 4 ds reads
    SGB(SGB_DSRD, 6);
    #pragma unroll
    for (int it = 0; it < 7; ++it) {
      SGB(SGB_WMMA, 2);
      SGB(SGB_DSRD, 4);
    }
    SGB(SGB_WMMA, 2);
  }

  // ---- epilogue: residual + skip accumulate -------------------------------
  #pragma unroll
  for (int nt = 0; nt < 8; ++nt) {
    const int n = nt * 16 + mrow;
    const float rb = res_b[n];
    const float sb = skip_b[n];
    #pragma unroll
    for (int r = 0; r < 8; ++r) {
      const size_t idx = (size_t)(r0 + r + mw) * CCH + n;
      xout[idx] = racc[nt][r] + rb + xin[idx];
      skip[idx] = skip[idx] + sacc[nt][r] + sb;
    }
  }
}

// ---- output head (linear => fold W0@W1 into one 128-vector) -----------------
__global__ void wn_fuse_head(const float* __restrict__ W0, const float* __restrict__ b0,
                             const float* __restrict__ W1, const float* __restrict__ b1,
                             float* __restrict__ v /* 129 floats */) {
  int c = threadIdx.x;   // 128 threads
  float acc = 0.f;
  for (int d = 0; d < CCH; ++d) acc += W0[c * CCH + d] * W1[d];
  v[c] = acc;
  if (c == 0) {
    float s = b1[0];
    for (int d = 0; d < CCH; ++d) s += b0[d] * W1[d];
    v[CCH] = s;
  }
}

__global__ void wn_head(const float* __restrict__ skip, const float* __restrict__ v,
                        float* __restrict__ out) {
  __shared__ float vs[CCH + 1];
  const int tid = threadIdx.x;
  if (tid < CCH + 1) vs[tid] = v[tid];
  __syncthreads();
  const int r = blockIdx.x * 256 + tid;
  const float4* row = (const float4*)(skip + (size_t)r * CCH);
  float acc = vs[CCH];
  #pragma unroll 4
  for (int c4 = 0; c4 < CCH / 4; ++c4) {
    float4 s4 = row[c4];
    acc += s4.x * vs[c4*4 + 0] + s4.y * vs[c4*4 + 1] +
           s4.z * vs[c4*4 + 2] + s4.w * vs[c4*4 + 3];
  }
  out[r] = acc;
}

// ---- driver -----------------------------------------------------------------
extern "C" void kernel_launch(void* const* d_in, const int* in_sizes, int n_in,
                              void* d_out, int out_size, void* d_ws, size_t ws_size,
                              hipStream_t stream) {
  const float* signal      = (const float*)d_in[0];
  const int*   timestep    = (const int*)  d_in[1];
  const float* in_W        = (const float*)d_in[2];
  const float* in_b        = (const float*)d_in[3];
  const float* emb_W0      = (const float*)d_in[4];
  const float* emb_b0      = (const float*)d_in[5];
  const float* emb_W1      = (const float*)d_in[6];
  const float* emb_b1      = (const float*)d_in[7];
  const float* blk_embed_W = (const float*)d_in[8];
  const float* blk_embed_b = (const float*)d_in[9];
  const float* blk_conv_W  = (const float*)d_in[10];
  const float* blk_conv_b  = (const float*)d_in[11];
  const float* blk_res_W   = (const float*)d_in[12];
  const float* blk_res_b   = (const float*)d_in[13];
  const float* blk_skip_W  = (const float*)d_in[14];
  const float* blk_skip_b  = (const float*)d_in[15];
  const float* out_W0      = (const float*)d_in[16];
  const float* out_b0      = (const float*)d_in[17];
  const float* out_W1      = (const float*)d_in[18];
  const float* out_b1      = (const float*)d_in[19];

  char* ws = (char*)d_ws;
  size_t off = 0;
  auto carve = [&](size_t bytes) -> char* {
    char* p = ws + off;
    off += (bytes + 255) & ~(size_t)255;
    return p;
  };
  float*     xA      = (float*)    carve((size_t)ROWS * CCH * 4);
  float*     xB      = (float*)    carve((size_t)ROWS * CCH * 4);
  float*     skipBuf = (float*)    carve((size_t)ROWS * CCH * 4);
  _Float16*  convT   = (_Float16*) carve((size_t)LAYERS * 3 * NCONV * KPAD * 2);
  _Float16*  resT    = (_Float16*) carve((size_t)LAYERS * CCH * KPAD * 2);
  _Float16*  skipT   = (_Float16*) carve((size_t)LAYERS * CCH * KPAD * 2);
  float*     embedB  = (float*)    carve((size_t)BATCH * 512 * 4);
  float*     eAll    = (float*)    carve((size_t)LAYERS * BATCH * CCH * 4);
  float*     vhead   = (float*)    carve((size_t)(CCH + 1) * 4);

  { // weight prep
    size_t tc = (size_t)LAYERS * 3 * NCONV * KPAD;
    wn_prep_conv<<<(unsigned)((tc + 255) / 256), 256, 0, stream>>>(blk_conv_W, convT);
    size_t t2 = (size_t)LAYERS * CCH * KPAD;
    wn_prep_cc<<<(unsigned)((t2 + 255) / 256), 256, 0, stream>>>(blk_res_W,  resT);
    wn_prep_cc<<<(unsigned)((t2 + 255) / 256), 256, 0, stream>>>(blk_skip_W, skipT);
  }
  wn_embed<<<1, 512, 0, stream>>>(timestep, emb_W0, emb_b0, emb_W1, emb_b1, embedB);
  wn_eall<<<(LAYERS * BATCH * CCH + 255) / 256, 256, 0, stream>>>(embedB, blk_embed_W,
                                                                  blk_embed_b, eAll);
  wn_init<<<(ROWS * CCH) / 256, 256, 0, stream>>>(signal, in_W, in_b, xA, skipBuf);
  wn_fuse_head<<<1, CCH, 0, stream>>>(out_W0, out_b0, out_W1, out_b1, vhead);

  float* xcur = xA;
  float* xnxt = xB;
  for (int i = 0; i < LAYERS; ++i) {
    const int dil = 1 << (i % 10);
    wn_layer<<<ROWS / 128, 256, 0, stream>>>(
        xcur, xnxt, skipBuf,
        convT + (size_t)i * 3 * NCONV * KPAD,
        resT  + (size_t)i * CCH * KPAD,
        skipT + (size_t)i * CCH * KPAD,
        blk_conv_b + (size_t)i * NCONV,
        blk_res_b  + (size_t)i * CCH,
        blk_skip_b + (size_t)i * CCH,
        eAll + (size_t)i * BATCH * CCH, dil);
    float* t = xcur; xcur = xnxt; xnxt = t;
  }

  wn_head<<<ROWS / 256, 256, 0, stream>>>(skipBuf, vhead, (float*)d_out);

  (void)in_sizes; (void)n_in; (void)out_size; (void)ws_size;
}
